// DeformableAttnBlock_21260088115212
// MI455X (gfx1250) — matrix operand order repaired and checked by reading, not compiled, not run on hardware
//
#include <hip/hip_runtime.h>
#include <hip/hip_bf16.h>

#define DEVFN __device__ __forceinline__

typedef __attribute__((ext_vector_type(16))) __bf16         v16bf;
typedef __attribute__((ext_vector_type(8)))  float          v8f;
typedef __attribute__((ext_vector_type(8)))  unsigned short v8u16;
typedef __attribute__((ext_vector_type(16))) unsigned short v16u16;

namespace {
constexpr int  B_  = 2;
constexpr int  T_  = 5;
constexpr int  C_  = 192;
constexpr int  NH_ = 4;
constexpr int  NL_ = 5;
constexpr int  NP_ = 4;
constexpr int  DH_ = 48;
constexpr int  H_  = 64;
constexpr int  W_  = 64;
constexpr int  HW_ = 4096;
constexpr long Lq_ = (long)T_ * HW_;   // 20480
}

// ---------------- helpers ----------------

DEVFN unsigned short f2bf(float f) {
  unsigned int u = __builtin_bit_cast(unsigned int, f);
  u += 0x7FFFu + ((u >> 16) & 1u);        // round-to-nearest-even
  return (unsigned short)(u >> 16);
}

DEVFN v16bf frag_zero() {
  v16u16 z = {};
  return __builtin_bit_cast(v16bf, z);
}

// A-fragment: 16x32 bf16 A matrix, lane holds row m, half h.
// Layout: e=0..15 -> K = h*16 + e (contiguous, matches frag_b).
DEVFN v16bf frag_a(const unsigned short* p, int h) {
  v8u16 lo = *reinterpret_cast<const v8u16*>(p + h * 16);
  v8u16 hi = *reinterpret_cast<const v8u16*>(p + h * 16 + 8);
  v16u16 c = __builtin_shufflevector(lo, hi, 0, 1, 2, 3, 4, 5, 6, 7,
                                     8, 9, 10, 11, 12, 13, 14, 15);
  return __builtin_bit_cast(v16bf, c);
}

// B-fragment: 32x16 bf16 B matrix, lane = column n (lane&15), half h.
// Element e -> K = h*16 + e : 16 contiguous bf16. p already includes h*16.
DEVFN v16bf frag_b(const unsigned short* p) {
  v8u16 lo = *reinterpret_cast<const v8u16*>(p);
  v8u16 hi = *reinterpret_cast<const v8u16*>(p + 8);
  v16u16 c = __builtin_shufflevector(lo, hi, 0, 1, 2, 3, 4, 5, 6, 7,
                                     8, 9, 10, 11, 12, 13, 14, 15);
  return __builtin_bit_cast(v16bf, c);
}

DEVFN v8f wmma_bf16(v16bf a, v16bf b, v8f c) {
  return __builtin_amdgcn_wmma_f32_16x16x32_bf16(
      /*neg_a=*/false, a, /*neg_b=*/false, b,
      /*c_mod=*/(short)0, c, /*reuse_a=*/false, /*reuse_b=*/false);
}

DEVFN float bil_border(const float* __restrict__ p, float x, float y) {
  x = fminf(fmaxf(x, 0.f), (float)(W_ - 1));
  y = fminf(fmaxf(y, 0.f), (float)(H_ - 1));
  float xf = floorf(x), yf = floorf(y);
  int x0 = (int)xf, y0 = (int)yf;
  int x1 = x0 + 1; if (x1 > W_ - 1) x1 = W_ - 1;
  int y1 = y0 + 1; if (y1 > H_ - 1) y1 = H_ - 1;
  float wx = x - xf, wy = y - yf;
  float v00 = p[y0 * W_ + x0], v01 = p[y0 * W_ + x1];
  float v10 = p[y1 * W_ + x0], v11 = p[y1 * W_ + x1];
  return v00 * (1.f - wx) * (1.f - wy) + v01 * wx * (1.f - wy) +
         v10 * (1.f - wx) * wy + v11 * wx * wy;
}

// ---------------- weight packing ----------------

// (O, I, 3, 3) f32  ->  (9, O, Ipad) bf16, zero padded in I.
__global__ void k_pack_conv_w(const float* __restrict__ w,
                              unsigned short* __restrict__ dst,
                              int O, int I, int Ipad) {
  long idx = (long)blockIdx.x * blockDim.x + threadIdx.x;
  long total = 9L * O * Ipad;
  if (idx >= total) return;
  int i = (int)(idx % Ipad);
  long r = idx / Ipad;
  int o = (int)(r % O);
  int tap = (int)(r / O);
  float v = 0.f;
  if (i < I) v = w[((long)o * I + i) * 9 + tap];
  dst[idx] = f2bf(v);
}

__global__ void k_pack_lin(const float* __restrict__ w,
                           unsigned short* __restrict__ dst, long n) {
  long i = (long)blockIdx.x * blockDim.x + threadIdx.x;
  if (i < n) dst[i] = f2bf(w[i]);
}

// ---------------- flow composition ----------------
// flows f32 [b][5][2][HW] = [ff02, ff12, 0, fb32, fb42]
__global__ void k_flows(const float* __restrict__ ffw,
                        const float* __restrict__ fbw,
                        float* __restrict__ flows) {
  long idx = (long)blockIdx.x * blockDim.x + threadIdx.x;
  if (idx >= (long)B_ * HW_) return;
  int pix = (int)(idx % HW_);
  int b = (int)(idx / HW_);
  int x = pix & 63, y = pix >> 6;
  const float* ff01 = ffw + ((long)b * 4 + 0) * 2 * HW_;
  const float* ff12 = ffw + ((long)b * 4 + 1) * 2 * HW_;
  const float* fb32 = fbw + ((long)b * 4 + 2) * 2 * HW_;
  const float* fb43 = fbw + ((long)b * 4 + 3) * 2 * HW_;
  float f01x = ff01[pix], f01y = ff01[HW_ + pix];
  float sx = (float)x + f01x, sy = (float)y + f01y;
  float o02x = f01x + bil_border(ff12, sx, sy);
  float o02y = f01y + bil_border(ff12 + HW_, sx, sy);
  float f43x = fb43[pix], f43y = fb43[HW_ + pix];
  float tx = (float)x + f43x, ty = (float)y + f43y;
  float o42x = f43x + bil_border(fb32, tx, ty);
  float o42y = f43y + bil_border(fb32 + HW_, tx, ty);
  float* F = flows + (long)b * NL_ * 2 * HW_;
  F[0 * 2 * HW_ + pix] = o02x;        F[0 * 2 * HW_ + HW_ + pix] = o02y;
  F[1 * 2 * HW_ + pix] = ff12[pix];   F[1 * 2 * HW_ + HW_ + pix] = ff12[HW_ + pix];
  F[2 * 2 * HW_ + pix] = 0.f;        F[2 * 2 * HW_ + HW_ + pix] = 0.f;
  F[3 * 2 * HW_ + pix] = fb32[pix];   F[3 * 2 * HW_ + HW_ + pix] = fb32[HW_ + pix];
  F[4 * 2 * HW_ + pix] = o42x;        F[4 * 2 * HW_ + HW_ + pix] = o42y;
}

// ---------------- build qk_in (bf16 HWC, 992 ch padded) ----------------
__global__ void k_build_qkin(const float* __restrict__ frame,
                             const float* __restrict__ ffw,
                             const float* __restrict__ fbw,
                             const float* __restrict__ flows,
                             unsigned short* __restrict__ qkin) {
  long idx = (long)blockIdx.x * blockDim.x + threadIdx.x;
  long total = (long)B_ * HW_ * 992;
  if (idx >= total) return;
  int c = (int)(idx % 992);
  long r = idx / 992;
  int pix = (int)(r % HW_);
  int b = (int)(r / HW_);
  float val = 0.f;
  if (c < 960) {
    int t = c / C_, cc = c % C_;
    const float* plane = frame + (((long)b * T_ + t) * C_ + cc) * HW_;
    if (t == 2) {
      val = plane[pix];
    } else {
      int x = pix & 63, y = pix >> 6;
      const float* F = flows + ((long)b * NL_ + t) * 2 * HW_;
      float fx = F[pix], fy = F[HW_ + pix];
      val = bil_border(plane, (float)x + fx, (float)y + fy);
    }
  } else if (c < 968) {
    val = ffw[((long)b * 8 + (c - 960)) * HW_ + pix];
  } else if (c < 976) {
    val = fbw[((long)b * 8 + (c - 968)) * HW_ + pix];
  }
  qkin[idx] = f2bf(val);
}

// frame (b,t,c,h,w) f32 -> bf16 HWC [b][HW][960]
__global__ void k_frame_hwc(const float* __restrict__ frame,
                            unsigned short* __restrict__ dst) {
  long idx = (long)blockIdx.x * blockDim.x + threadIdx.x;
  long total = (long)B_ * HW_ * 960;
  if (idx >= total) return;
  int c = (int)(idx % 960);
  long r = idx / 960;
  int pix = (int)(r % HW_);
  int b = (int)(r / HW_);
  dst[idx] = f2bf(frame[((long)b * 960 + c) * HW_ + pix]);
}

// ---------------- implicit-GEMM 3x3 conv via WMMA ----------------
// Grid: x = H/4 rows (4 waves/block, one row each), y = Cout/32, z = nimg.
// Each wave: 32 couts (2 M-tiles) x one full 64-pixel row (4 N-tiles):
// 8 independent accumulators, 2 A-loads + 4 B-loads per 8 wmmas.
// act:  bf16 [nimg][HW][Cinp]   (HWC, Cinp multiple of 32)
// wpk:  bf16 [9][Cout][Cinp]
// out*: token-major [nimg*(Cout/192)][HW][192]
__global__ __launch_bounds__(128) void k_conv_wmma(
    const unsigned short* __restrict__ act,
    const unsigned short* __restrict__ wpk,
    const float* __restrict__ bias,
    unsigned short* __restrict__ out_bf,
    float* __restrict__ out_f32,
    int Cinp, int Cout, int dil, int leaky) {
  const int lane = (int)(threadIdx.x & 31);
  const int wave = (int)(threadIdx.x >> 5);
  const int y    = (int)blockIdx.x * 4 + wave;   // image row
  const int cb   = (int)blockIdx.y;              // 32-wide cout block
  const int img  = (int)blockIdx.z;
  const int m = lane & 15;
  const int h = lane >> 4;
  v8f acc[8];
#pragma unroll
  for (int i = 0; i < 8; ++i) acc[i] = {};
  const int kbn = Cinp >> 5;
  for (int tap = 0; tap < 9; ++tap) {
    const int dy = (tap / 3 - 1) * dil;
    const int dx = (tap % 3 - 1) * dil;
    const int yy = y + dy;
    const bool rowok = (yy >= 0) && (yy < H_);
    const unsigned short* pA0 = wpk + ((long)tap * Cout + cb * 32 + m) * Cinp;
    const unsigned short* pA1 = pA0 + (long)16 * Cinp;
    const long rowbase = (long)img * HW_ + (long)yy * W_;
    const unsigned short* pB[4];
    bool val[4];
#pragma unroll
    for (int s = 0; s < 4; ++s) {
      int xx = s * 16 + m + dx;
      val[s] = rowok && (xx >= 0) && (xx < W_);
      pB[s] = act + (rowbase + xx) * Cinp + h * 16;
    }
    for (int kb = 0; kb < kbn; ++kb) {
      v16bf a0 = frag_a(pA0 + kb * 32, h);
      v16bf a1 = frag_a(pA1 + kb * 32, h);
      v16bf b0 = val[0] ? frag_b(pB[0] + kb * 32) : frag_zero();
      v16bf b1 = val[1] ? frag_b(pB[1] + kb * 32) : frag_zero();
      v16bf b2 = val[2] ? frag_b(pB[2] + kb * 32) : frag_zero();
      v16bf b3 = val[3] ? frag_b(pB[3] + kb * 32) : frag_zero();
      acc[0] = wmma_bf16(a0, b0, acc[0]);
      acc[1] = wmma_bf16(a0, b1, acc[1]);
      acc[2] = wmma_bf16(a0, b2, acc[2]);
      acc[3] = wmma_bf16(a0, b3, acc[3]);
      acc[4] = wmma_bf16(a1, b0, acc[4]);
      acc[5] = wmma_bf16(a1, b1, acc[5]);
      acc[6] = wmma_bf16(a1, b2, acc[6]);
      acc[7] = wmma_bf16(a1, b3, acc[7]);
    }
  }
  // D layout: cout = cb*32 + ctt*16 + r + 8*h ; pixel = y*64 + s*16 + (lane&15)
#pragma unroll
  for (int ctt = 0; ctt < 2; ++ctt) {
    const int c0 = cb * 32 + ctt * 16 + h * 8;
    const int t = c0 / C_;
    const int cc = c0 % C_;
    const long pbase = ((long)img * (Cout / C_) + t) * HW_ + (long)y * W_ + m;
#pragma unroll
    for (int s = 0; s < 4; ++s) {
      const long obase = (pbase + s * 16) * C_ + cc;
      float v[8];
#pragma unroll
      for (int r = 0; r < 8; ++r) {
        float xv = acc[ctt * 4 + s][r] + bias[c0 + r];
        if (leaky) xv = (xv >= 0.f) ? xv : 0.1f * xv;
        v[r] = xv;
      }
      if (out_f32) {
#pragma unroll
        for (int r = 0; r < 8; ++r) out_f32[obase + r] = v[r];
      }
      if (out_bf) {
#pragma unroll
        for (int r = 0; r < 8; ++r) out_bf[obase + r] = f2bf(v[r]);
      }
    }
  }
}

// ---------------- token GEMM via WMMA: out = X * Wt^T + bias ----------------
// Grid: x = Ntok/64/4 (4 waves/block, 64 tokens each), y = Cout/16.
// X: bf16 [Ntok][K] ; Wt: bf16 [Cout][K] ; out token-major [Ntok][Cout]
__global__ __launch_bounds__(128) void k_gemm_tok(
    const unsigned short* __restrict__ X,
    const unsigned short* __restrict__ Wt,
    const float* __restrict__ bias,
    unsigned short* __restrict__ out_bf,
    float* __restrict__ out_f32,
    int K, int Cout) {
  const int lane = (int)(threadIdx.x & 31);
  const int wave = (int)(threadIdx.x >> 5);
  const long tg = (long)blockIdx.x * 4 + wave;   // group of 64 tokens
  const int ct = (int)blockIdx.y;
  const int m = lane & 15;
  const int h = lane >> 4;
  const unsigned short* pA = Wt + (long)(ct * 16 + m) * K;
  const unsigned short* pB0 = X + (tg * 64 + 0 * 16 + m) * K + h * 16;
  const unsigned short* pB1 = X + (tg * 64 + 1 * 16 + m) * K + h * 16;
  const unsigned short* pB2 = X + (tg * 64 + 2 * 16 + m) * K + h * 16;
  const unsigned short* pB3 = X + (tg * 64 + 3 * 16 + m) * K + h * 16;
  v8f acc0 = {}, acc1 = {}, acc2 = {}, acc3 = {};
  const int kbn = K >> 5;
  for (int kb = 0; kb < kbn; ++kb) {
    v16bf a = frag_a(pA + kb * 32, h);
    v16bf b0 = frag_b(pB0 + kb * 32);
    v16bf b1 = frag_b(pB1 + kb * 32);
    v16bf b2 = frag_b(pB2 + kb * 32);
    v16bf b3 = frag_b(pB3 + kb * 32);
    acc0 = wmma_bf16(a, b0, acc0);
    acc1 = wmma_bf16(a, b1, acc1);
    acc2 = wmma_bf16(a, b2, acc2);
    acc3 = wmma_bf16(a, b3, acc3);
  }
  const int c0 = ct * 16 + h * 8;
  const v8f accs[4] = {acc0, acc1, acc2, acc3};
#pragma unroll
  for (int s = 0; s < 4; ++s) {
    long tok = tg * 64 + s * 16 + m;
    long obase = tok * (long)Cout + c0;
#pragma unroll
    for (int r = 0; r < 8; ++r) {
      float xv = accs[s][r] + bias[c0 + r];
      if (out_f32) out_f32[obase + r] = xv;
      if (out_bf) out_bf[obase + r] = f2bf(xv);
    }
  }
}

// ---------------- softmax over NL*NP = 20 ----------------
__global__ void k_softmax(float* __restrict__ aw, long n) {
  long i = (long)blockIdx.x * blockDim.x + threadIdx.x;
  if (i >= n) return;
  float* p = aw + i * 20;
  float mx = p[0];
#pragma unroll
  for (int j = 1; j < 20; ++j) mx = fmaxf(mx, p[j]);
  float s = 0.f;
#pragma unroll
  for (int j = 0; j < 20; ++j) { float e = __expf(p[j] - mx); p[j] = e; s += e; }
  float inv = 1.f / s;
#pragma unroll
  for (int j = 0; j < 20; ++j) p[j] *= inv;
}

// ---------------- deformable sampling + weighted accumulation ----------------
__global__ void k_deform(const float* __restrict__ vproj,   // [B][Lq][192]
                         const float* __restrict__ offs,    // [B][Lq][160]
                         const float* __restrict__ aw,      // [B][Lq][80]
                         const float* __restrict__ flows,   // [B][5][2][HW]
                         unsigned short* __restrict__ attn_tok) {
  long idx = (long)blockIdx.x * blockDim.x + threadIdx.x;
  long total = (long)B_ * Lq_ * NH_ * DH_;
  if (idx >= total) return;
  int d = (int)(idx % DH_);
  long r = idx / DH_;
  int nh = (int)(r % NH_);
  r /= NH_;
  long q = r % Lq_;
  int b = (int)(r / Lq_);
  int pix = (int)(q & (HW_ - 1));
  int x = pix & 63, y = pix >> 6;
  float rx = ((float)x + 0.5f) * (1.f / 64.f);
  float ry = ((float)y + 0.5f) * (1.f / 64.f);
  const float* offp = offs + ((long)b * Lq_ + q) * 160 + nh * 40;
  const float* awp = aw + ((long)b * Lq_ + q) * 80 + nh * 20;
  const float* Fb = flows + (long)b * NL_ * 2 * HW_;
  int ch = nh * DH_ + d;
  float acc = 0.f;
  for (int l = 0; l < NL_; ++l) {
    float fx = Fb[(l * 2 + 0) * HW_ + pix] * (1.f / 64.f);
    float fy = Fb[(l * 2 + 1) * HW_ + pix] * (1.f / 64.f);
    const float* vpl = vproj + ((long)b * Lq_ + (long)l * HW_) * C_;
    for (int p = 0; p < NP_; ++p) {
      float lx = rx + offp[(l * 4 + p) * 2 + 0] * (1.f / 64.f) + fx;
      float ly = ry + offp[(l * 4 + p) * 2 + 1] * (1.f / 64.f) + fy;
      float sx = lx * 64.f - 0.5f;
      float sy = ly * 64.f - 0.5f;
      float xf = floorf(sx), yf = floorf(sy);
      int ix = (int)xf, iy = (int)yf;
      float wx = sx - xf, wy = sy - yf;
      float s = 0.f;
#pragma unroll
      for (int cy = 0; cy < 2; ++cy) {
#pragma unroll
        for (int cx = 0; cx < 2; ++cx) {
          int xi = ix + cx, yi = iy + cy;
          if (xi >= 0 && xi < W_ && yi >= 0 && yi < H_) {
            float wgt = (cx ? wx : 1.f - wx) * (cy ? wy : 1.f - wy);
            s += vpl[((long)yi * W_ + xi) * C_ + ch] * wgt;
          }
        }
      }
      acc += s * awp[l * 4 + p];
    }
  }
  attn_tok[((long)b * Lq_ + q) * C_ + ch] = f2bf(acc);
}

// ---------------- residual + seq_in build ----------------
__global__ void k_residual(const float* __restrict__ ffout,
                           const float* __restrict__ frame,
                           const float* __restrict__ srcframe,
                           float* __restrict__ outres,
                           unsigned short* __restrict__ seqin) {
  long idx = (long)blockIdx.x * blockDim.x + threadIdx.x;
  long total = (long)B_ * T_ * HW_ * C_;
  if (idx >= total) return;
  int c = (int)(idx % C_);
  long r = idx / C_;
  int pix = (int)(r % HW_);
  int img = (int)(r / HW_);
  long nchw = ((long)img * C_ + c) * HW_ + pix;
  float v = ffout[idx] + frame[nchw];
  outres[idx] = v;
  long sbase = ((long)img * HW_ + pix) * (2 * C_);
  seqin[sbase + c] = f2bf(v);
  seqin[sbase + C_ + c] = f2bf(srcframe[nchw]);
}

// ---------------- final: out = outres + ffn2 (to NCHW) + srcframe copy ----------------
__global__ void k_final(const float* __restrict__ outres,
                        const float* __restrict__ ffn2,
                        const float* __restrict__ srcframe,
                        float* __restrict__ dout) {
  long idx = (long)blockIdx.x * blockDim.x + threadIdx.x;
  long total = (long)B_ * T_ * HW_ * C_;
  if (idx >= total) return;
  int c = (int)(idx % C_);
  long r = idx / C_;
  int pix = (int)(r % HW_);
  int img = (int)(r / HW_);
  long nchw = ((long)img * C_ + c) * HW_ + pix;
  dout[nchw] = outres[idx] + ffn2[idx];
  dout[total + nchw] = srcframe[nchw];
}

// ---------------- host ----------------

extern "C" void kernel_launch(void* const* d_in, const int* in_sizes, int n_in,
                              void* d_out, int out_size, void* d_ws, size_t ws_size,
                              hipStream_t stream) {
  (void)in_sizes; (void)n_in; (void)out_size; (void)ws_size;
  const float* frame    = (const float*)d_in[0];
  const float* srcframe = (const float*)d_in[1];
  const float* ffw      = (const float*)d_in[2];
  const float* fbw      = (const float*)d_in[3];
  const float* w_qk     = (const float*)d_in[4];
  const float* b_qk     = (const float*)d_in[5];
  const float* w_v      = (const float*)d_in[6];
  const float* b_v      = (const float*)d_in[7];
  const float* w_vproj  = (const float*)d_in[8];
  const float* b_vproj  = (const float*)d_in[9];
  const float* w_off    = (const float*)d_in[10];
  const float* b_off    = (const float*)d_in[11];
  const float* w_attn   = (const float*)d_in[12];
  const float* b_attn   = (const float*)d_in[13];
  const float* w_out    = (const float*)d_in[14];
  const float* b_out    = (const float*)d_in[15];
  const float* w_ff     = (const float*)d_in[16];
  const float* b_ff     = (const float*)d_in[17];
  const float* w_ffn1   = (const float*)d_in[18];
  const float* b_ffn1   = (const float*)d_in[19];
  const float* w_ffn2   = (const float*)d_in[20];
  const float* b_ffn2   = (const float*)d_in[21];

  char* ws = (char*)d_ws;
  size_t off = 0;
  auto alloc = [&](size_t bytes) -> char* {
    off = (off + 255) & ~(size_t)255;
    char* p = ws + off;
    off += bytes;
    return p;
  };
  typedef unsigned short u16;

  float* flows  = (float*)alloc((size_t)B_ * NL_ * 2 * HW_ * 4);
  u16* qkin     = (u16*)alloc((size_t)B_ * HW_ * 992 * 2);
  u16* fhwc     = (u16*)alloc((size_t)B_ * HW_ * 960 * 2);
  u16* wqk_p    = (u16*)alloc((size_t)9 * 960 * 992 * 2);
  u16* wv_p     = (u16*)alloc((size_t)9 * 960 * 960 * 2);
  u16* wff_p    = (u16*)alloc((size_t)9 * 192 * 192 * 2);
  u16* wffn1_p  = (u16*)alloc((size_t)9 * 192 * 384 * 2);
  u16* wffn2_p  = (u16*)alloc((size_t)9 * 192 * 192 * 2);
  u16* wvproj_p = (u16*)alloc((size_t)192 * 192 * 2);
  u16* woff_p   = (u16*)alloc((size_t)160 * 192 * 2);
  u16* wattn_p  = (u16*)alloc((size_t)80 * 192 * 2);
  u16* wout_p   = (u16*)alloc((size_t)192 * 192 * 2);
  u16* q_tok    = (u16*)alloc((size_t)B_ * Lq_ * C_ * 2);
  u16* v_tok    = (u16*)alloc((size_t)B_ * Lq_ * C_ * 2);
  float* vproj  = (float*)alloc((size_t)B_ * Lq_ * C_ * 4);
  float* offsb  = (float*)alloc((size_t)B_ * Lq_ * 160 * 4);
  float* awb    = (float*)alloc((size_t)B_ * Lq_ * 80 * 4);
  u16* attn_tok = (u16*)alloc((size_t)B_ * Lq_ * C_ * 2);
  u16* wouttok  = (u16*)alloc((size_t)B_ * Lq_ * C_ * 2);
  float* ffout  = (float*)alloc((size_t)B_ * T_ * HW_ * C_ * 4);
  float* outres = (float*)alloc((size_t)B_ * T_ * HW_ * C_ * 4);
  u16* seqin    = (u16*)alloc((size_t)B_ * T_ * HW_ * 2 * C_ * 2);
  u16* ffn1o    = (u16*)alloc((size_t)B_ * T_ * HW_ * C_ * 2);
  float* ffn2o  = (float*)alloc((size_t)B_ * T_ * HW_ * C_ * 4);

  const int TB = 256;
  auto g1 = [](long n) { return (unsigned)((n + 255) / 256); };

  // weight packing
  k_pack_conv_w<<<g1(9L * 960 * 992), TB, 0, stream>>>(w_qk, wqk_p, 960, 976, 992);
  k_pack_conv_w<<<g1(9L * 960 * 960), TB, 0, stream>>>(w_v, wv_p, 960, 960, 960);
  k_pack_conv_w<<<g1(9L * 192 * 192), TB, 0, stream>>>(w_ff, wff_p, 192, 192, 192);
  k_pack_conv_w<<<g1(9L * 192 * 384), TB, 0, stream>>>(w_ffn1, wffn1_p, 192, 384, 384);
  k_pack_conv_w<<<g1(9L * 192 * 192), TB, 0, stream>>>(w_ffn2, wffn2_p, 192, 192, 192);
  k_pack_lin<<<g1(192 * 192), TB, 0, stream>>>(w_vproj, wvproj_p, 192 * 192);
  k_pack_lin<<<g1(160 * 192), TB, 0, stream>>>(w_off, woff_p, 160 * 192);
  k_pack_lin<<<g1(80 * 192), TB, 0, stream>>>(w_attn, wattn_p, 80 * 192);
  k_pack_lin<<<g1(192 * 192), TB, 0, stream>>>(w_out, wout_p, 192 * 192);

  // flows + inputs in bf16 HWC
  k_flows<<<g1((long)B_ * HW_), TB, 0, stream>>>(ffw, fbw, flows);
  k_build_qkin<<<g1((long)B_ * HW_ * 992), TB, 0, stream>>>(frame, ffw, fbw, flows, qkin);
  k_frame_hwc<<<g1((long)B_ * HW_ * 960), TB, 0, stream>>>(frame, fhwc);

  // conv grid: x = 16 (64 rows / 4 waves), y = Cout/32, z = nimg
  // queries / value convs (leaky)
  k_conv_wmma<<<dim3(H_ / 4, 960 / 32, B_), 128, 0, stream>>>(
      qkin, wqk_p, b_qk, q_tok, nullptr, 992, 960, 1, 1);
  k_conv_wmma<<<dim3(H_ / 4, 960 / 32, B_), 128, 0, stream>>>(
      fhwc, wv_p, b_v, v_tok, nullptr, 960, 960, 1, 1);

  const long Ntok = (long)B_ * Lq_;           // 40960 tokens
  const unsigned gx = (unsigned)(Ntok / 64 / 4);  // 160 blocks of 4 waves
  k_gemm_tok<<<dim3(gx, 192 / 16), 128, 0, stream>>>(v_tok, wvproj_p, b_vproj, nullptr, vproj, 192, 192);
  k_gemm_tok<<<dim3(gx, 160 / 16), 128, 0, stream>>>(q_tok, woff_p, b_off, nullptr, offsb, 192, 160);
  k_gemm_tok<<<dim3(gx, 80 / 16), 128, 0, stream>>>(q_tok, wattn_p, b_attn, nullptr, awb, 192, 80);

  k_softmax<<<g1(Ntok * NH_), TB, 0, stream>>>(awb, Ntok * NH_);
  k_deform<<<g1((long)B_ * Lq_ * NH_ * DH_), TB, 0, stream>>>(vproj, offsb, awb, flows, attn_tok);

  k_gemm_tok<<<dim3(gx, 192 / 16), 128, 0, stream>>>(attn_tok, wout_p, b_out, wouttok, nullptr, 192, 192);

  // ff conv (no activation), residual, ffn
  k_conv_wmma<<<dim3(H_ / 4, 192 / 32, B_ * T_), 128, 0, stream>>>(
      wouttok, wff_p, b_ff, nullptr, ffout, 192, 192, 1, 0);
  k_residual<<<g1((long)B_ * T_ * HW_ * C_), TB, 0, stream>>>(ffout, frame, srcframe, outres, seqin);
  k_conv_wmma<<<dim3(H_ / 4, 192 / 32, B_ * T_), 128, 0, stream>>>(
      seqin, wffn1_p, b_ffn1, ffn1o, nullptr, 384, 192, 2, 1);
  k_conv_wmma<<<dim3(H_ / 4, 192 / 32, B_ * T_), 128, 0, stream>>>(
      ffn1o, wffn2_p, b_ffn2, nullptr, ffn2o, 192, 192, 1, 0);

  k_final<<<g1((long)B_ * T_ * HW_ * C_), TB, 0, stream>>>(outres, ffn2o, srcframe, (float*)d_out);
}
